// ExphormerAttention_10763188043963
// MI455X (gfx1250) — matrix operand order repaired and checked by reading, not compiled, not run on hardware
//
#include <hip/hip_runtime.h>
#include <hip/hip_bf16.h>
#include <math.h>

#define N_NODES   100000
#define N_EDGES   800000
#define IN_DIM    128
#define OUT_DIM   128
#define NUM_HEADS 8
#define HEAD_DIM  16

typedef __attribute__((ext_vector_type(16))) __bf16    v16bf;
typedef __attribute__((ext_vector_type(8)))  float     v8f;
typedef __attribute__((ext_vector_type(8)))  unsigned  v8u;

// Pack two f32 into a dword of two bf16 (truncation): low half from f0, high from f1.
static __device__ inline unsigned pack_hi(float f0, float f1) {
    unsigned u0 = __builtin_bit_cast(unsigned, f0);
    unsigned u1 = __builtin_bit_cast(unsigned, f1);
    return (u0 >> 16) | (u1 & 0xffff0000u);
}
// Residual (f - bf16(f)) packed the same way: recovers ~16 mantissa bits total.
static __device__ inline unsigned pack_lo(float f0, float f1) {
    unsigned u0 = __builtin_bit_cast(unsigned, f0);
    unsigned u1 = __builtin_bit_cast(unsigned, f1);
    float r0 = f0 - __builtin_bit_cast(float, u0 & 0xffff0000u);
    float r1 = f1 - __builtin_bit_cast(float, u1 & 0xffff0000u);
    return (__builtin_bit_cast(unsigned, r0) >> 16) |
           (__builtin_bit_cast(unsigned, r1) & 0xffff0000u);
}

// ---------------------------------------------------------------------------
// Kernel 0 (one-time prep, 96 waves): pre-split W{q,k,v} into bf16 hi/lo
// fragments laid out exactly as WMMA B registers.
// Layout (v8u units): [(mat*8 + coltile)*4 + kk)*2 + hilo]*32 + lane
// B layout used here (contiguous k, same as A): col n = lane&15; v8u element v packs k = 16*g + 2v, +1.
// ---------------------------------------------------------------------------
__global__ __launch_bounds__(256) void pack_w_kernel(
    const float* __restrict__ Wq, const float* __restrict__ Wk,
    const float* __restrict__ Wv, v8u* __restrict__ Wpack)
{
    const int t       = blockIdx.x * 256 + threadIdx.x;
    const int wave_id = t >> 5;          // 0..95
    const int lane    = t & 31;
    const int mat     = wave_id >> 5;            // 0..2
    const int rem     = wave_id & 31;
    const int wtile   = rem >> 2;                // 0..7 column tile
    const int kk      = rem & 3;                 // 0..3 k-step
    const int g       = lane >> 4;
    const int nl      = lane & 15;

    const float* W = (mat == 0) ? Wq : (mat == 1) ? Wk : Wv;
    const int n = wtile * 16 + nl;

    v8u bh, bl;
    #pragma unroll
    for (int v = 0; v < 8; ++v) {
        const int k = kk * 32 + 16 * g + 2 * v;
        const float f0 = W[(size_t)k       * OUT_DIM + n];
        const float f1 = W[(size_t)(k + 1) * OUT_DIM + n];
        bh[v] = pack_hi(f0, f1);
        bl[v] = pack_lo(f0, f1);
    }
    const size_t base = (size_t)(((mat * 8 + wtile) * 4 + kk) * 2) * 32 + lane;
    Wpack[base]      = bh;   // hilo = 0
    Wpack[base + 32] = bl;   // hilo = 1
}

// ---------------------------------------------------------------------------
// Kernel 1: Q/K/V = x @ W{q,k,v} + b via WMMA (bf16 hi/lo split, f32 accum).
// Block = 16 node rows x 256 threads (8 waves). Wave w -> output cols 16w..16w+15.
// B fragments arrive pre-packed; A fragments built once, reused for 3 matrices.
// ---------------------------------------------------------------------------
__global__ __launch_bounds__(256) void qkv_gemm_kernel(
    const float* __restrict__ x, const v8u* __restrict__ Wpack,
    const float* __restrict__ bq, const float* __restrict__ bk,
    const float* __restrict__ bv,
    float* __restrict__ Q, float* __restrict__ K, float* __restrict__ V)
{
    __shared__ float xs[16 * IN_DIM];   // 8 KB tile of x

    const int r0  = blockIdx.x * 16;
    const int tid = threadIdx.x;

    // Cooperative tile load: 2048 floats = 512 float4, 256 threads -> 2 each.
    {
        const float4* src = (const float4*)(x + (size_t)r0 * IN_DIM);
        float4* dst = (float4*)xs;
        dst[tid]       = src[tid];
        dst[tid + 256] = src[tid + 256];
    }
    __syncthreads();

    const int wave = tid >> 5;     // 0..7 -> column tile
    const int lane = tid & 31;
    const int g    = lane >> 4;    // lane group
    const int nl   = lane & 15;
    const int n0   = wave * 16;

    // A fragments (hi/lo) for 4 K-steps.  A k order used here (matches B):
    // row m = lane&15; packed dword v holds k = 16g + 2v, +1.
    v16bf Ah[4], Al[4];
    {
        const int m = nl;
        #pragma unroll
        for (int kk = 0; kk < 4; ++kk) {
            const float* xr = xs + m * IN_DIM + kk * 32 + 16 * g;
            v8u ah, al;
            #pragma unroll
            for (int v = 0; v < 8; ++v) {
                const int kb = 2 * v;
                const float f0 = xr[kb];
                const float f1 = xr[kb + 1];
                ah[v] = pack_hi(f0, f1);
                al[v] = pack_lo(f0, f1);
            }
            Ah[kk] = __builtin_bit_cast(v16bf, ah);
            Al[kk] = __builtin_bit_cast(v16bf, al);
        }
    }

    const float* bs[3] = {bq, bk, bv};
    float*       Ys[3] = {Q,  K,  V};

    #pragma unroll
    for (int mat = 0; mat < 3; ++mat) {
        v8f c = {};
        #pragma unroll
        for (int kk = 0; kk < 4; ++kk) {
            const size_t base =
                (size_t)(((mat * 8 + wave) * 4 + kk) * 2) * 32 + lane;
            const v16bf Bh = __builtin_bit_cast(v16bf, Wpack[base]);
            const v16bf Bl = __builtin_bit_cast(v16bf, Wpack[base + 32]);
            c = __builtin_amdgcn_wmma_f32_16x16x32_bf16(false, Ah[kk], false, Bh,
                                                        (short)0, c, false, false);
            c = __builtin_amdgcn_wmma_f32_16x16x32_bf16(false, Ah[kk], false, Bl,
                                                        (short)0, c, false, false);
            c = __builtin_amdgcn_wmma_f32_16x16x32_bf16(false, Al[kk], false, Bh,
                                                        (short)0, c, false, false);
        }
        // C/D layout: col n = lane&15, row m = v + 8*lanegroup
        const float bias = bs[mat][n0 + nl];
        float* Y = Ys[mat] + (size_t)r0 * OUT_DIM + n0 + nl;
        #pragma unroll
        for (int v = 0; v < 8; ++v)
            Y[(size_t)(v + 8 * g) * OUT_DIM] = c[v] + bias;
    }
}

// ---------------------------------------------------------------------------
// Kernel 2: per-(edge, head) attention. Gathers hit L2 (QKV = 153.6 MB < 192 MB).
// ---------------------------------------------------------------------------
__global__ __launch_bounds__(256) void edge_attn_kernel(
    const int* __restrict__ ei,     // [2, E] (JAX default demotes int64->int32)
    const float* __restrict__ Q, const float* __restrict__ K,
    const float* __restrict__ V,
    float* __restrict__ wV, float* __restrict__ Z)
{
    const long long idx = (long long)blockIdx.x * blockDim.x + threadIdx.x;
    if (idx >= (long long)N_EDGES * NUM_HEADS) return;
    const int e = (int)(idx >> 3);
    const int h = (int)(idx & 7);

    const int src = ei[e];
    const int dst = ei[N_EDGES + e];

    const float4* kp = (const float4*)(K + (size_t)src * OUT_DIM + h * HEAD_DIM);
    const float4* qp = (const float4*)(Q + (size_t)dst * OUT_DIM + h * HEAD_DIM);
    const float4* vp = (const float4*)(V + (size_t)src * OUT_DIM + h * HEAD_DIM);
    __builtin_prefetch(vp, 0, 0);   // global_prefetch_b8: overlap V gather with dot

    float dot = 0.0f;
    #pragma unroll
    for (int i = 0; i < 4; ++i) {
        const float4 a = kp[i], b = qp[i];
        dot += a.x * b.x + a.y * b.y + a.z * b.z + a.w * b.w;
    }
    dot *= 0.25f;                              // 1/sqrt(16)
    dot = fminf(5.0f, fmaxf(-5.0f, dot));
    const float s = __expf(dot);

    float* op = wV + (size_t)dst * OUT_DIM + h * HEAD_DIM;
    #pragma unroll
    for (int i = 0; i < 4; ++i) {
        const float4 v4 = vp[i];
        unsafeAtomicAdd(op + 4 * i + 0, v4.x * s);
        unsafeAtomicAdd(op + 4 * i + 1, v4.y * s);
        unsafeAtomicAdd(op + 4 * i + 2, v4.z * s);
        unsafeAtomicAdd(op + 4 * i + 3, v4.w * s);
    }
    unsafeAtomicAdd(Z + (size_t)dst * NUM_HEADS + h, s);
}

// ---------------------------------------------------------------------------
// Kernel 3: h_out = wV / (Z + 1e-6), in place on d_out.
// ---------------------------------------------------------------------------
__global__ __launch_bounds__(256) void normalize_kernel(
    float* __restrict__ wV, const float* __restrict__ Z)
{
    const int idx = blockIdx.x * blockDim.x + threadIdx.x;  // node*H + h
    if (idx >= N_NODES * NUM_HEADS) return;
    const float r = 1.0f / (Z[idx] + 1e-6f);
    float4* p = (float4*)(wV + (size_t)idx * HEAD_DIM);
    #pragma unroll
    for (int i = 0; i < 4; ++i) {
        float4 v = p[i];
        v.x *= r; v.y *= r; v.z *= r; v.w *= r;
        p[i] = v;
    }
}

extern "C" void kernel_launch(void* const* d_in, const int* in_sizes, int n_in,
                              void* d_out, int out_size, void* d_ws, size_t ws_size,
                              hipStream_t stream) {
    const float* x  = (const float*)d_in[0];
    const int*   ei = (const int*)  d_in[1];
    // d_in[2] = virt_h, d_in[3] = virt_edge_index: unused dummies
    const float* Wq = (const float*)d_in[4];
    const float* bq = (const float*)d_in[5];
    const float* Wk = (const float*)d_in[6];
    const float* bk = (const float*)d_in[7];
    const float* Wv = (const float*)d_in[8];
    const float* bv = (const float*)d_in[9];
    float* out = (float*)d_out;

    const size_t qkv_elems = (size_t)N_NODES * OUT_DIM;
    float* Q = (float*)d_ws;
    float* K = Q + qkv_elems;
    float* V = K + qkv_elems;
    float* Z = V + qkv_elems;
    v8u* Wpack = (v8u*)(Z + (size_t)N_NODES * NUM_HEADS);  // 192 KB, 32B aligned

    // Accumulators must start at zero every launch (harness poisons buffers).
    hipMemsetAsync(out, 0, qkv_elems * sizeof(float), stream);
    hipMemsetAsync(Z, 0, (size_t)N_NODES * NUM_HEADS * sizeof(float), stream);

    pack_w_kernel<<<12, 256, 0, stream>>>(Wq, Wk, Wv, Wpack);

    qkv_gemm_kernel<<<N_NODES / 16, 256, 0, stream>>>(
        x, Wpack, bq, bk, bv, Q, K, V);

    const long long pairs = (long long)N_EDGES * NUM_HEADS;
    edge_attn_kernel<<<(int)((pairs + 255) / 256), 256, 0, stream>>>(
        ei, Q, K, V, out, Z);

    normalize_kernel<<<(N_NODES * NUM_HEADS + 255) / 256, 256, 0, stream>>>(out, Z);
}
